// TopologyNetwork_12214886990283
// MI455X (gfx1250) — matrix operand order repaired and checked by reading, not compiled, not run on hardware
//
#include <hip/hip_runtime.h>

#define NODE_DIM 128
#define HIDDEN 64
#define N_NODES 2048

typedef __attribute__((ext_vector_type(16))) _Float16 v16h;
typedef __attribute__((ext_vector_type(8)))  float    v8f;

static __device__ __forceinline__ v8f wmma_f16(v16h a, v16h b, v8f c) {
  // (neg_a, A, neg_b, B, c_mod, C, reuse_a, reuse_b)
  return __builtin_amdgcn_wmma_f32_16x16x32_f16(false, a, false, b, (short)0, c,
                                                false, false);
}

// One wave (32 lanes) computes a 16-row slab:
//   h = X(16x128) @ Wenc(128x64) + benc          (WMMA, 4 k-steps x 4 n-tiles)
//   a = h @ Wa ; c = h @ Wb + b1                 (WMMA, 2 k-steps x 4 n-tiles)
__global__ __launch_bounds__(32)
void topo_encoder_kernel(const float* __restrict__ X,
                         const float* __restrict__ Wenc,
                         const float* __restrict__ benc,
                         const float* __restrict__ W1,
                         const float* __restrict__ b1,
                         float* __restrict__ aout,
                         float* __restrict__ cout) {
  const int lane = threadIdx.x;
  const int lm   = lane & 15;   // M (A) / N (B,C/D)
  const int lh   = lane >> 4;   // lane-half selector

  const int r0 = blockIdx.x * 16;

  // ---- A fragments of X: 16x128 f16, 4 k-steps of 32 ----
  // A k order used here (contiguous, same as B): element e -> K = e + 16*lh ; row M = lm
  v16h Af[4];
#pragma unroll
  for (int kk = 0; kk < 4; ++kk) {
#pragma unroll
    for (int e = 0; e < 16; ++e) {
      const int K = kk * 32 + e + (lh << 4);
      Af[kk][e] = (_Float16)X[(r0 + lm) * NODE_DIM + K];
    }
  }

  // ---- h = X @ Wenc + benc ----
  // B k order used here (contiguous, same as A): element e -> K = e + 16*lh ; col N = lm
  v8f hacc[4];
#pragma unroll
  for (int nn = 0; nn < 4; ++nn) {
    v8f acc = {0.f, 0.f, 0.f, 0.f, 0.f, 0.f, 0.f, 0.f};
#pragma unroll
    for (int kk = 0; kk < 4; ++kk) {
      v16h Bf;
#pragma unroll
      for (int e = 0; e < 16; ++e) {
        const int K = kk * 32 + e + (lh << 4);
        Bf[e] = (_Float16)Wenc[K * HIDDEN + nn * 16 + lm];
      }
      acc = wmma_f16(Af[kk], Bf, acc);
    }
    const float bias = benc[nn * 16 + lm];
#pragma unroll
    for (int q = 0; q < 8; ++q) acc[q] += bias;
    hacc[nn] = acc;
  }

  // ---- relayout h (C/D frag -> A frag) through LDS as f16 ----
  __shared__ _Float16 hlds[16][HIDDEN];
#pragma unroll
  for (int nn = 0; nn < 4; ++nn)
#pragma unroll
    for (int q = 0; q < 8; ++q)
      hlds[q + 8 * lh][nn * 16 + lm] = (_Float16)hacc[nn][q];
  __syncthreads();

  v16h hAf[2];
#pragma unroll
  for (int kk = 0; kk < 2; ++kk)
#pragma unroll
    for (int e = 0; e < 16; ++e)
      hAf[kk][e] = hlds[lm][kk * 32 + e + (lh << 4)];

  // ---- a = h @ Wa ; c = h @ Wb + b1 ----
#pragma unroll
  for (int nn = 0; nn < 4; ++nn) {
    v8f aacc = {0.f, 0.f, 0.f, 0.f, 0.f, 0.f, 0.f, 0.f};
    v8f cacc = {0.f, 0.f, 0.f, 0.f, 0.f, 0.f, 0.f, 0.f};
#pragma unroll
    for (int kk = 0; kk < 2; ++kk) {
      v16h BfA, BfB;
#pragma unroll
      for (int e = 0; e < 16; ++e) {
        const int K = kk * 32 + e + (lh << 4);
        BfA[e] = (_Float16)W1[K * HIDDEN + nn * 16 + lm];            // Wa
        BfB[e] = (_Float16)W1[(HIDDEN + K) * HIDDEN + nn * 16 + lm]; // Wb
      }
      aacc = wmma_f16(hAf[kk], BfA, aacc);
      cacc = wmma_f16(hAf[kk], BfB, cacc);
    }
    const float bias1 = b1[nn * 16 + lm];
#pragma unroll
    for (int q = 0; q < 8; ++q) {
      const int row = r0 + q + 8 * lh;
      aout[row * HIDDEN + nn * 16 + lm] = aacc[q];
      cout[row * HIDDEN + nn * 16 + lm] = cacc[q] + bias1;
    }
  }
}

// Pairwise kernel: out[i,j] = sigmoid( sum_h relu(a[i,h]+c[j,h]) * W2[h] + b2 )
// 128(i) x 64(j) tile per 256-thread block; 8x4 outputs per thread.
// Scalar add/max/fma (3 VALU per h per pair) — histogram-verified better than
// the packed VOP3P variant (which forced cross-asm register packing moves),
// and leaves the backend free to form VOPD duals.
#define BI 128
#define BJ 64
#define LDSS 68  // padded row stride (floats): 16B aligned, bank-conflict-free

__global__ __launch_bounds__(256)
void topo_pair_kernel(const float* __restrict__ A,
                      const float* __restrict__ C,
                      const float* __restrict__ W2,
                      const float* __restrict__ b2,
                      float* __restrict__ out) {
  __shared__ float als[BI * LDSS];
  __shared__ float cls[BJ * LDSS];

  const int tid = threadIdx.x;
  const int tx  = tid & 15;   // j lane
  const int ty  = tid >> 4;   // i lane
  const int i0  = blockIdx.y * BI;
  const int j0  = blockIdx.x * BJ;

  // stage a-tile (128x64) and c-tile (64x64) into LDS, float4 at a time
#pragma unroll
  for (int k = 0; k < 8; ++k) {
    const int idx = tid + 256 * k;            // 0..2047
    const int row = idx >> 4, c4 = (idx & 15) << 2;
    *(float4*)&als[row * LDSS + c4] =
        *(const float4*)&A[(i0 + row) * HIDDEN + c4];
  }
#pragma unroll
  for (int k = 0; k < 4; ++k) {
    const int idx = tid + 256 * k;            // 0..1023
    const int row = idx >> 4, c4 = (idx & 15) << 2;
    *(float4*)&cls[row * LDSS + c4] =
        *(const float4*)&C[(j0 + row) * HIDDEN + c4];
  }
  __syncthreads();

  float acc[8][4];
#pragma unroll
  for (int u = 0; u < 8; ++u)
#pragma unroll
    for (int v = 0; v < 4; ++v) acc[u][v] = 0.f;

#pragma unroll 1
  for (int h = 0; h < HIDDEN; h += 4) {
    const float4 w = *(const float4*)(W2 + h);  // uniform -> s_load
    float4 ar[8], cr[4];
#pragma unroll
    for (int u = 0; u < 8; ++u)
      ar[u] = *(const float4*)&als[(ty + 16 * u) * LDSS + h];
#pragma unroll
    for (int v = 0; v < 4; ++v)
      cr[v] = *(const float4*)&cls[(tx + 16 * v) * LDSS + h];
#pragma unroll
    for (int u = 0; u < 8; ++u) {
#pragma unroll
      for (int v = 0; v < 4; ++v) {
        const float t0 = fmaxf(ar[u].x + cr[v].x, 0.f);
        const float t1 = fmaxf(ar[u].y + cr[v].y, 0.f);
        const float t2 = fmaxf(ar[u].z + cr[v].z, 0.f);
        const float t3 = fmaxf(ar[u].w + cr[v].w, 0.f);
        acc[u][v] += t0 * w.x + t1 * w.y + t2 * w.z + t3 * w.w;
      }
    }
  }

  const float bias2 = b2[0];
#pragma unroll
  for (int u = 0; u < 8; ++u) {
    const int gi = i0 + ty + 16 * u;
#pragma unroll
    for (int v = 0; v < 4; ++v) {
      const int gj = j0 + tx + 16 * v;
      const float x = acc[u][v] + bias2;
      const float T = 1.0f / (1.0f + __expf(-x));
      out[gi * N_NODES + gj] = (gi == gj) ? 0.f : T;
    }
  }
}

extern "C" void kernel_launch(void* const* d_in, const int* in_sizes, int n_in,
                              void* d_out, int out_size, void* d_ws, size_t ws_size,
                              hipStream_t stream) {
  const float* X    = (const float*)d_in[0];  // node_features (2048,128)
  const float* Wenc = (const float*)d_in[1];  // (128,64)
  const float* benc = (const float*)d_in[2];  // (64,)
  const float* W1   = (const float*)d_in[3];  // (128,64)
  const float* b1   = (const float*)d_in[4];  // (64,)
  const float* W2   = (const float*)d_in[5];  // (64,1)
  const float* b2   = (const float*)d_in[6];  // (1,)
  float* out = (float*)d_out;

  float* a_ws = (float*)d_ws;                   // 2048*64 f32
  float* c_ws = a_ws + N_NODES * HIDDEN;        // 2048*64 f32

  topo_encoder_kernel<<<N_NODES / 16, 32, 0, stream>>>(X, Wenc, benc, W1, b1,
                                                       a_ws, c_ws);

  dim3 grid(N_NODES / BJ, N_NODES / BI);  // (32,16)
  topo_pair_kernel<<<grid, 256, 0, stream>>>(a_ws, c_ws, W2, b2, out);
}
